// GINConvNet_1013612282049
// MI455X (gfx1250) — compile-verified, matrix operand order checked by reading, not run on hardware
//
#include <hip/hip_runtime.h>

// ---------------------------------------------------------------------------
// GINConvNet forward for MI455X (gfx1250, wave32, WMMA bf16 path)
// B operands stored K-major ([N][K]) so both A and B fragments load as
// packed 32-bit K-pairs; 16x32 output tile per wave (2 WMMAs / K-step).
// ---------------------------------------------------------------------------

#define N_NODES   65536
#define N_EDGES   2097152
#define NB        1024
#define FXD       78
#define DIM       32
#define EMBD      128
#define OUTD      128
#define PLEN      1000
#define KS        8
#define CLEN      121      // EMB - KS + 1
#define NF        32
#define VOCAB     26
#define GK        208      // VOCAB * KS
#define GKP       224      // padded to multiple of 32
#define XTK       3872     // NF * CLEN

typedef __attribute__((ext_vector_type(8)))  float  v8f;
typedef __attribute__((ext_vector_type(16))) __bf16 v16bf;

union BF16Frag { v16bf v; unsigned short u[16]; unsigned int w[8]; };

__device__ __forceinline__ unsigned short f32_to_bf16(float f) {
  unsigned int u = __float_as_uint(f);
  unsigned int r = u + 0x7FFFu + ((u >> 16) & 1u);   // round-to-nearest-even
  return (unsigned short)(r >> 16);
}
__device__ __forceinline__ float bf16_to_f32(unsigned short h) {
  return __uint_as_float(((unsigned int)h) << 16);
}
// base K of halfword pair vg (0..7) for lane-half `half` in a 16x32 fragment
__device__ __forceinline__ int pair_base(int vg, int half) {
  return (vg < 4) ? (half * 8 + 2 * vg) : (16 + half * 8 + 2 * (vg - 4));
}

// ---------------------------------------------------------------- utilities
__global__ void zero_kernel(float* p, long long n) {
  long long i = (long long)blockIdx.x * blockDim.x + threadIdx.x;
  long long s = (long long)gridDim.x * blockDim.x;
  for (; i < n; i += s) p[i] = 0.f;
}
__global__ void copy_kernel(const float* __restrict__ src, float* __restrict__ dst, long long n) {
  long long i = (long long)blockIdx.x * blockDim.x + threadIdx.x;
  long long s = (long long)gridDim.x * blockDim.x;
  for (; i < n; i += s) dst[i] = src[i];
}
// f32 [K x N] row-major  ->  bf16 [N x K] (K-major for WMMA B fragments)
__global__ void cvt_bf16_t_kernel(const float* __restrict__ src, unsigned short* __restrict__ dst,
                                  int K, int Nn) {
  long long i = (long long)blockIdx.x * blockDim.x + threadIdx.x;
  long long s = (long long)gridDim.x * blockDim.x;
  long long tot = (long long)K * Nn;
  for (; i < tot; i += s) {
    int k = (int)(i / Nn);
    int n = (int)(i - (long long)k * Nn);
    dst[(size_t)n * K + k] = f32_to_bf16(src[i]);
  }
}

// ------------------------------------------------------------- GIN: edges
__global__ void edge_scatter_kernel(const float* __restrict__ h, const int* __restrict__ src,
                                    const int* __restrict__ dst, float* __restrict__ agg,
                                    long long total, int D) {
  long long i = (long long)blockIdx.x * blockDim.x + threadIdx.x;
  long long s = (long long)gridDim.x * blockDim.x;
  for (; i < total; i += s) {
    int e = (int)(i / D);
    int d = (int)(i - (long long)e * D);
    atomicAdd(&agg[(size_t)dst[e] * D + d], h[(size_t)src[e] * D + d]);
  }
}

// ---------------------------------------------- GIN: MLP + ReLU + BN stats
__global__ __launch_bounds__(256)
void gin_mlp_kernel(const float* __restrict__ agg, int D1,
                    const float* __restrict__ wi, const float* __restrict__ bi,
                    const float* __restrict__ wo, const float* __restrict__ bo,
                    float* __restrict__ h2, float* __restrict__ stats) {
  __shared__ float swi[FXD * DIM];
  __shared__ float swo[DIM * DIM];
  __shared__ float ssum[DIM];
  __shared__ float ssq[DIM];
  int tid = threadIdx.x;
  for (int i = tid; i < D1 * DIM; i += 256) swi[i] = wi[i];
  for (int i = tid; i < DIM * DIM; i += 256) swo[i] = wo[i];
  if (tid < DIM) { ssum[tid] = 0.f; ssq[tid] = 0.f; }
  __syncthreads();

  int node = blockIdx.x * 256 + tid;
  float hid[DIM];
#pragma unroll
  for (int o = 0; o < DIM; ++o) hid[o] = bi[o];
  const float* arow = agg + (size_t)node * D1;
  for (int d = 0; d < D1; ++d) {
    float ad = arow[d];
#pragma unroll
    for (int o = 0; o < DIM; ++o) hid[o] += ad * swi[d * DIM + o];
  }
#pragma unroll
  for (int o = 0; o < DIM; ++o) hid[o] = fmaxf(hid[o], 0.f);

  float* hrow = h2 + (size_t)node * DIM;
#pragma unroll
  for (int o2 = 0; o2 < DIM; ++o2) {
    float s = bo[o2];
#pragma unroll
    for (int o = 0; o < DIM; ++o) s += hid[o] * swo[o * DIM + o2];
    s = fmaxf(s, 0.f);
    hrow[o2] = s;
    atomicAdd(&ssum[o2], s);
    atomicAdd(&ssq[o2], s * s);
  }
  __syncthreads();
  if (tid < DIM) {
    atomicAdd(&stats[tid], ssum[tid]);
    atomicAdd(&stats[DIM + tid], ssq[tid]);
  }
}

// ----------------------------------------------------------- GIN: BN apply
__global__ void bn_kernel(const float* __restrict__ h2, const float* __restrict__ stats,
                          const float* __restrict__ gamma, const float* __restrict__ beta,
                          float* __restrict__ h) {
  long long i = (long long)blockIdx.x * blockDim.x + threadIdx.x;
  long long s = (long long)gridDim.x * blockDim.x;
  const float invN = 1.f / (float)N_NODES;
  for (; i < (long long)N_NODES * DIM; i += s) {
    int d = (int)(i & (DIM - 1));
    float m = stats[d] * invN;
    float var = stats[DIM + d] * invN - m * m;
    h[i] = gamma[d] * (h2[i] - m) * rsqrtf(var + 1e-5f) + beta[d];
  }
}

// --------------------------------------------------------- global add pool
__global__ void pool_kernel(const float* __restrict__ h, const int* __restrict__ batch,
                            float* __restrict__ hg) {
  long long i = (long long)blockIdx.x * blockDim.x + threadIdx.x;
  long long s = (long long)gridDim.x * blockDim.x;
  for (; i < (long long)N_NODES * DIM; i += s) {
    int n = (int)(i >> 5);
    int d = (int)(i & 31);
    atomicAdd(&hg[(size_t)batch[n] * DIM + d], h[i]);
  }
}

// ------------------------------------------- xd = ReLU(hg@Wd+b) -> xc[:,0:128]
__global__ __launch_bounds__(128)
void xd_kernel(const float* __restrict__ hg, const float* __restrict__ w,
               const float* __restrict__ bias, unsigned short* __restrict__ xc) {
  __shared__ float sh[DIM];
  int b = blockIdx.x, o = threadIdx.x;
  if (o < DIM) sh[o] = hg[b * DIM + o];
  __syncthreads();
  float s = bias[o];
#pragma unroll
  for (int d = 0; d < DIM; ++d) s += sh[d] * w[d * OUTD + o];
  xc[(size_t)b * 256 + o] = f32_to_bf16(fmaxf(s, 0.f));
}

// ----------- Em_t[j][kk] = emb[kk/8][j + kk%8]  (bf16, K-major, zero padded)
__global__ __launch_bounds__(224)
void em_build_kernel(const float* __restrict__ emb, unsigned short* __restrict__ EmT) {
  int j = blockIdx.x;     // 0..127 (output column)
  int kk = threadIdx.x;   // 0..223 (K index)
  float v = 0.f;
  if (kk < GK && j < CLEN) v = emb[(kk >> 3) * EMBD + j + (kk & 7)];
  EmT[(size_t)j * GKP + kk] = f32_to_bf16(v);
}

// ---------------------------------------------------------------------------
// Protein conv restructured:  c[b] (32x121) = Gm_b (32x208) @ Em (208x128)
//   Gm_b[f, v*8+k] = sum_{p: target[b,p]==v} conv_w[f,p,k]  (built in LDS,
//   each thread owns one (f,k) column -> no atomics), converted to a bf16
//   row-major LDS tile, then bf16 WMMA GEMM against K-major Em.
// ---------------------------------------------------------------------------
__global__ __launch_bounds__(256)
void conv_gemm_kernel(const int* __restrict__ target, const float* __restrict__ conv_w,
                      const float* __restrict__ conv_b, const unsigned short* __restrict__ EmT,
                      unsigned short* __restrict__ cmat) {
  __shared__ float Gs[VOCAB * 256];          // Gs[v][f*8+k]
  __shared__ unsigned short Gb[NF * GKP];    // bf16 A tile, [f][kk]
  int b = blockIdx.x;
  int tid = threadIdx.x;
#pragma unroll
  for (int v = 0; v < VOCAB; ++v) Gs[v * 256 + tid] = 0.f;

  int f = tid >> 3, kq = tid & 7;
  const float* wp = conv_w + (size_t)f * (PLEN * KS) + kq;
  const int* tb = target + (size_t)b * PLEN;
  for (int p = 0; p < PLEN; ++p) {
    int v = tb[p];
    Gs[v * 256 + tid] += wp[p * KS];
  }
  __syncthreads();
  for (int i = tid; i < NF * GKP; i += 256) {
    int m = i / GKP, kk = i - m * GKP;
    float gv = (kk < GK) ? Gs[(kk >> 3) * 256 + m * 8 + (kk & 7)] : 0.f;
    Gb[i] = f32_to_bf16(gv);
  }
  __syncthreads();

  int wave = tid >> 5, lane = tid & 31;
  int half = lane >> 4, mn = lane & 15;
  int tmi = wave >> 2;                 // 0..1  (M tile)
  int tnp = wave & 3;                  // 0..3  (N tile pair)
  const unsigned short* Arow = Gb + (tmi * 16 + mn) * GKP;
  const unsigned short* B0 = EmT + (size_t)(tnp * 32 + mn) * GKP;
  const unsigned short* B1 = B0 + (size_t)16 * GKP;

  v8f acc0 = {0.f, 0.f, 0.f, 0.f, 0.f, 0.f, 0.f, 0.f};
  v8f acc1 = {0.f, 0.f, 0.f, 0.f, 0.f, 0.f, 0.f, 0.f};
  for (int kb = 0; kb < GKP; kb += 32) {
    BF16Frag a, b0, b1;
#pragma unroll
    for (int vg = 0; vg < 8; ++vg) {
      int k = kb + pair_base(vg, half);
      a.w[vg]  = *(const unsigned int*)(Arow + k);
      b0.w[vg] = *(const unsigned int*)(B0 + k);
      b1.w[vg] = *(const unsigned int*)(B1 + k);
    }
    acc0 = __builtin_amdgcn_wmma_f32_16x16x32_bf16(false, a.v, false, b0.v,
                                                   (short)0, acc0, false, false);
    acc1 = __builtin_amdgcn_wmma_f32_16x16x32_bf16(false, a.v, false, b1.v,
                                                   (short)0, acc1, false, false);
  }
#pragma unroll
  for (int r = 0; r < 8; ++r) {
    int fo = tmi * 16 + r + 8 * half;
    int j0 = tnp * 32 + mn;
    int j1 = j0 + 16;
    if (j0 < CLEN)
      cmat[(size_t)b * XTK + fo * CLEN + j0] = f32_to_bf16(acc0[r] + conv_b[fo]);
    if (j1 < CLEN)
      cmat[(size_t)b * XTK + fo * CLEN + j1] = f32_to_bf16(acc1[r] + conv_b[fo]);
  }
}

// ---------------------------------------------------------------------------
// Generic bf16 GEMM: out_bf16[M x N] = act(A[MxK] @ B + bias)
// A row-major [M x K]; Bt K-major [N x K]. One 16x32 tile per wave32,
// two v_wmma_f32_16x16x32_bf16 per K-step sharing the A fragment.
// M mult of 16, N mult of 32, K mult of 32.
// ---------------------------------------------------------------------------
__global__ __launch_bounds__(256)
void gemm_bf16_kernel(const unsigned short* __restrict__ A, int lda,
                      const unsigned short* __restrict__ Bt,
                      unsigned short* __restrict__ outB, int ldo, int colOff,
                      const float* __restrict__ bias,
                      int M, int Nn, int K, int relu) {
  int wave = threadIdx.x >> 5;
  int lane = threadIdx.x & 31;
  int tilesNp = Nn >> 5;
  int tileId = blockIdx.x * 8 + wave;
  if (tileId >= (M >> 4) * tilesNp) return;    // whole-wave exit, EXEC stays full
  int tm = tileId / tilesNp, tnp = tileId % tilesNp;
  int half = lane >> 4, mn = lane & 15;
  const unsigned short* Arow = A + (size_t)(tm * 16 + mn) * lda;
  const unsigned short* B0 = Bt + (size_t)(tnp * 32 + mn) * K;
  const unsigned short* B1 = B0 + (size_t)16 * K;

  v8f acc0 = {0.f, 0.f, 0.f, 0.f, 0.f, 0.f, 0.f, 0.f};
  v8f acc1 = {0.f, 0.f, 0.f, 0.f, 0.f, 0.f, 0.f, 0.f};
  for (int kb = 0; kb < K; kb += 32) {
    BF16Frag a, b0, b1;
#pragma unroll
    for (int vg = 0; vg < 8; ++vg) {
      int k = kb + pair_base(vg, half);
      a.w[vg]  = *(const unsigned int*)(Arow + k);
      b0.w[vg] = *(const unsigned int*)(B0 + k);
      b1.w[vg] = *(const unsigned int*)(B1 + k);
    }
    acc0 = __builtin_amdgcn_wmma_f32_16x16x32_bf16(false, a.v, false, b0.v,
                                                   (short)0, acc0, false, false);
    acc1 = __builtin_amdgcn_wmma_f32_16x16x32_bf16(false, a.v, false, b1.v,
                                                   (short)0, acc1, false, false);
  }
#pragma unroll
  for (int r = 0; r < 8; ++r) {
    int m = tm * 16 + r + 8 * half;
    int n0 = tnp * 32 + mn;
    float v0 = acc0[r] + bias[n0];
    float v1 = acc1[r] + bias[n0 + 16];
    if (relu) { v0 = fmaxf(v0, 0.f); v1 = fmaxf(v1, 0.f); }
    outB[(size_t)m * ldo + colOff + n0] = f32_to_bf16(v0);
    outB[(size_t)m * ldo + colOff + n0 + 16] = f32_to_bf16(v1);
  }
}

// ------------------------------------------------- final: out[b] = h2b.w_out+b
__global__ __launch_bounds__(256)
void out_kernel(const unsigned short* __restrict__ h2b, const float* __restrict__ w_out,
                const float* __restrict__ b_out, float* __restrict__ out) {
  __shared__ float red[256];
  int b = blockIdx.x, tid = threadIdx.x;
  red[tid] = bf16_to_f32(h2b[(size_t)b * 256 + tid]) * w_out[tid];
  __syncthreads();
  for (int s = 128; s > 0; s >>= 1) {
    if (tid < s) red[tid] += red[tid + s];
    __syncthreads();
  }
  if (tid == 0) out[b] = red[0] + b_out[0];
}

// ---------------------------------------------------------------------------
extern "C" void kernel_launch(void* const* d_in, const int* in_sizes, int n_in,
                              void* d_out, int out_size, void* d_ws, size_t ws_size,
                              hipStream_t stream) {
  (void)in_sizes; (void)n_in; (void)out_size; (void)ws_size;
  const float* x       = (const float*)d_in[0];
  const int*   ei      = (const int*)d_in[1];
  const int*   batch   = (const int*)d_in[2];
  const int*   target  = (const int*)d_in[3];
  const float* w1a     = (const float*)d_in[4];
  const float* b1a     = (const float*)d_in[5];
  const float* w1b     = (const float*)d_in[6];
  const float* b1b     = (const float*)d_in[7];
  const float* wa      = (const float*)d_in[8];
  const float* ba      = (const float*)d_in[9];
  const float* wb      = (const float*)d_in[10];
  const float* bb      = (const float*)d_in[11];
  const float* gamma   = (const float*)d_in[12];
  const float* beta    = (const float*)d_in[13];
  const float* w_fcxd  = (const float*)d_in[14];
  const float* b_fcxd  = (const float*)d_in[15];
  const float* emb     = (const float*)d_in[16];
  const float* conv_w  = (const float*)d_in[17];
  const float* conv_b  = (const float*)d_in[18];
  const float* w_fcxt  = (const float*)d_in[19];
  const float* b_fcxt  = (const float*)d_in[20];
  const float* w_fc1   = (const float*)d_in[21];
  const float* b_fc1   = (const float*)d_in[22];
  const float* w_fc2   = (const float*)d_in[23];
  const float* b_fc2   = (const float*)d_in[24];
  const float* w_out   = (const float*)d_in[25];
  const float* b_out   = (const float*)d_in[26];
  const int* esrc = ei;
  const int* edst = ei + N_EDGES;

  char* ws = (char*)d_ws;
  // ---- persistent region -------------------------------------------------
  const size_t AGG_OFF  = 0;                              // N*78 f32 (20,447,232 B)
  const size_t AGG_SZ   = (size_t)N_NODES * FXD * 4;
  const size_t HBUF_OFF = AGG_OFF + AGG_SZ;               // N*32 f32
  const size_t HBUF_SZ  = (size_t)N_NODES * DIM * 4;
  const size_t H2_OFF   = HBUF_OFF + HBUF_SZ;             // N*32 f32
  const size_t ST_OFF   = H2_OFF + HBUF_SZ;               // 64 f32
  const size_t HG_OFF   = ST_OFF + 256;                   // 1024*32 f32
  const size_t XC_OFF   = HG_OFF + (size_t)NB * DIM * 4;  // 1024*256 bf16
  // ---- protein-stage scratch aliased into dead AGG region ----------------
  const size_t EM_OFF   = AGG_OFF;                        // 128*224 bf16 (K-major)
  const size_t CM_OFF   = EM_OFF + 65536;                 // 1024*3872 bf16
  const size_t WXT_OFF  = CM_OFF + (size_t)NB * XTK * 2;  // [128][3872] bf16
  const size_t WF1_OFF  = WXT_OFF + (size_t)XTK * OUTD * 2;  // [1024][256] bf16
  const size_t WF2_OFF  = WF1_OFF + (size_t)256 * 1024 * 2;  // [256][1024] bf16
  const size_t H1B_OFF  = WF2_OFF + (size_t)1024 * 256 * 2;  // 1024*1024 bf16
  const size_t H2B_OFF  = H1B_OFF + (size_t)1024 * 1024 * 2; // 1024*256 bf16

  float* agg   = (float*)(ws + AGG_OFF);
  float* hbuf  = (float*)(ws + HBUF_OFF);
  float* h2buf = (float*)(ws + H2_OFF);
  float* stats = (float*)(ws + ST_OFF);
  float* hg    = (float*)(ws + HG_OFF);
  unsigned short* xcb  = (unsigned short*)(ws + XC_OFF);
  unsigned short* EmT  = (unsigned short*)(ws + EM_OFF);
  unsigned short* cmat = (unsigned short*)(ws + CM_OFF);
  unsigned short* wxtb = (unsigned short*)(ws + WXT_OFF);
  unsigned short* wf1b = (unsigned short*)(ws + WF1_OFF);
  unsigned short* wf2b = (unsigned short*)(ws + WF2_OFF);
  unsigned short* h1b  = (unsigned short*)(ws + H1B_OFF);
  unsigned short* h2b  = (unsigned short*)(ws + H2B_OFF);

  // ================= GIN stage (memory/atomic bound -> VALU) =============
  for (int L = 0; L < 5; ++L) {
    const float* hin = (L == 0) ? x : hbuf;
    int D1 = (L == 0) ? FXD : DIM;
    const float* wi = (L == 0) ? w1a : wa + (size_t)(L - 1) * DIM * DIM;
    const float* bi = (L == 0) ? b1a : ba + (size_t)(L - 1) * DIM;
    const float* wo = (L == 0) ? w1b : wb + (size_t)(L - 1) * DIM * DIM;
    const float* bo = (L == 0) ? b1b : bb + (size_t)(L - 1) * DIM;
    long long nAgg = (long long)N_NODES * D1;
    copy_kernel<<<4096, 256, 0, stream>>>(hin, agg, nAgg);
    edge_scatter_kernel<<<8192, 256, 0, stream>>>(hin, esrc, edst, agg,
                                                  (long long)N_EDGES * D1, D1);
    zero_kernel<<<1, 64, 0, stream>>>(stats, 64);
    gin_mlp_kernel<<<N_NODES / 256, 256, 0, stream>>>(agg, D1, wi, bi, wo, bo, h2buf, stats);
    bn_kernel<<<8192, 256, 0, stream>>>(h2buf, stats, gamma + L * DIM, beta + L * DIM, hbuf);
  }
  zero_kernel<<<128, 256, 0, stream>>>(hg, (long long)NB * DIM);
  pool_kernel<<<8192, 256, 0, stream>>>(hbuf, batch, hg);
  xd_kernel<<<NB, 128, 0, stream>>>(hg, w_fcxd, b_fcxd, xcb);   // xc[:, 0:128]

  // ================= protein branch (WMMA bf16) ==========================
  em_build_kernel<<<EMBD, GKP, 0, stream>>>(emb, EmT);
  conv_gemm_kernel<<<NB, 256, 0, stream>>>(target, conv_w, conv_b, EmT, cmat);

  cvt_bf16_t_kernel<<<1937, 256, 0, stream>>>(w_fcxt, wxtb, XTK, OUTD);
  // xt = cmat @ w_fcxt + b_fcxt  -> xc[:, 128:256]
  gemm_bf16_kernel<<<(64 * 4) / 8, 256, 0, stream>>>(cmat, XTK, wxtb,
                                                     xcb, 256, 128, b_fcxt,
                                                     NB, OUTD, XTK, 0);

  // ================= joint head (WMMA bf16) ==============================
  cvt_bf16_t_kernel<<<1024, 256, 0, stream>>>(w_fc1, wf1b, 256, 1024);
  gemm_bf16_kernel<<<(64 * 32) / 8, 256, 0, stream>>>(xcb, 256, wf1b,
                                                      h1b, 1024, 0, b_fc1,
                                                      NB, 1024, 256, 1);
  cvt_bf16_t_kernel<<<1024, 256, 0, stream>>>(w_fc2, wf2b, 1024, 256);
  gemm_bf16_kernel<<<(64 * 8) / 8, 256, 0, stream>>>(h1b, 1024, wf2b,
                                                     h2b, 256, 0, b_fc2,
                                                     NB, 256, 1024, 1);
  out_kernel<<<NB, 256, 0, stream>>>(h2b, w_out, b_out, (float*)d_out);
}
